// PreProcess_65240553226800
// MI455X (gfx1250) — matrix operand order repaired and checked by reading, not compiled, not run on hardware
//
#include <hip/hip_runtime.h>
#include <climits>
#include <cstdint>

#define NPTS   262144
#define VOX    0.05f
#define HSENT  0x7fffffff
#define K3BLK  128

typedef __attribute__((ext_vector_type(16))) _Float16 v16h;
typedef __attribute__((ext_vector_type(8)))  float    v8f;

__device__ __forceinline__ unsigned hmix(int h) {
  unsigned x = (unsigned)h * 2654435761u;
  x ^= x >> 16;
  return x;
}

// ---------------- init / zero ----------------
__global__ void k_zero_f32(float* p, size_t n) {
  size_t i = (size_t)blockIdx.x * blockDim.x + threadIdx.x;
  size_t stride = (size_t)gridDim.x * blockDim.x;
  for (; i < n; i += stride) p[i] = 0.0f;
}

__global__ void k_init(int* keys, int* vals, size_t tn, int* qmm, int B) {
  size_t i = (size_t)blockIdx.x * blockDim.x + threadIdx.x;
  size_t stride = (size_t)gridDim.x * blockDim.x;
  for (size_t k = i; k < tn; k += stride) { keys[k] = -1; vals[k] = HSENT; }
  if (i < (size_t)B * 6) qmm[i] = ((i % 6) < 3) ? INT_MAX : INT_MIN;
}

// ---------------- per-batch qmin/qmax over valid points ----------------
__global__ void k_minmax(const float* pcds, int* qmm, int N) {
  int b = blockIdx.y;
  int i = blockIdx.x * 256 + threadIdx.x;
  const float* p = pcds + ((size_t)b * N + i) * 3;
  float px = p[0], py = p[1], pz = p[2];
  bool valid = (px != 0.0f) || (py != 0.0f) || (pz != 0.0f);
  int q0 = (int)rintf(px / VOX), q1 = (int)rintf(py / VOX), q2 = (int)rintf(pz / VOX);
  int mn0 = valid ? q0 : INT_MAX, mn1 = valid ? q1 : INT_MAX, mn2 = valid ? q2 : INT_MAX;
  int mx0 = valid ? q0 : INT_MIN, mx1 = valid ? q1 : INT_MIN, mx2 = valid ? q2 : INT_MIN;
  for (int o = 16; o > 0; o >>= 1) {       // wave32 reduction
    mn0 = min(mn0, __shfl_down(mn0, o));
    mn1 = min(mn1, __shfl_down(mn1, o));
    mn2 = min(mn2, __shfl_down(mn2, o));
    mx0 = max(mx0, __shfl_down(mx0, o));
    mx1 = max(mx1, __shfl_down(mx1, o));
    mx2 = max(mx2, __shfl_down(mx2, o));
  }
  __shared__ int s[8][6];
  int lane = threadIdx.x & 31, wid = threadIdx.x >> 5;
  if (lane == 0) {
    s[wid][0] = mn0; s[wid][1] = mn1; s[wid][2] = mn2;
    s[wid][3] = mx0; s[wid][4] = mx1; s[wid][5] = mx2;
  }
  __syncthreads();
  if (threadIdx.x == 0) {
    int a0 = s[0][0], a1 = s[0][1], a2 = s[0][2];
    int c0 = s[0][3], c1 = s[0][4], c2 = s[0][5];
    for (int w = 1; w < 8; ++w) {
      a0 = min(a0, s[w][0]); a1 = min(a1, s[w][1]); a2 = min(a2, s[w][2]);
      c0 = max(c0, s[w][3]); c1 = max(c1, s[w][4]); c2 = max(c2, s[w][5]);
    }
    atomicMin(&qmm[b * 6 + 0], a0); atomicMin(&qmm[b * 6 + 1], a1); atomicMin(&qmm[b * 6 + 2], a2);
    atomicMax(&qmm[b * 6 + 3], c0); atomicMax(&qmm[b * 6 + 4], c1); atomicMax(&qmm[b * 6 + 5], c2);
  }
}

// ---------------- ravel hash + first-occurrence insert (atomicMin idx) ----------------
__global__ void k_hash_insert(const float* pcds, const int* qmm, int* hbuf,
                              int* keys, int* vals, int cap, int N) {
  int b = blockIdx.y;
  int i = blockIdx.x * 256 + threadIdx.x;
  const float* p = pcds + ((size_t)b * N + i) * 3;
  __builtin_prefetch(p + 3 * 256, 0, 1);   // global_prefetch_b8, next tile
  float px = p[0], py = p[1], pz = p[2];
  bool valid = (px != 0.0f) || (py != 0.0f) || (pz != 0.0f);
  int h = HSENT;
  if (valid) {
    int x0 = (int)rintf(px / VOX) - qmm[b * 6 + 0];
    int x1 = (int)rintf(py / VOX) - qmm[b * 6 + 1];
    int x2 = (int)rintf(pz / VOX) - qmm[b * 6 + 2];
    int X1 = qmm[b * 6 + 4] - qmm[b * 6 + 1] + 1;
    int X2 = qmm[b * 6 + 5] - qmm[b * 6 + 2] + 1;
    h = (x0 * X1 + x1) * X2 + x2;
    int* K = keys + (size_t)b * cap;
    int* V = vals + (size_t)b * cap;
    unsigned slot = hmix(h) & (unsigned)(cap - 1);
    for (int t = 0; t < cap; ++t) {
      int k = atomicCAS(&K[slot], -1, h);
      if (k == -1 || k == h) { atomicMin(&V[slot], i); break; }
      slot = (slot + 1) & (unsigned)(cap - 1);
    }
  }
  hbuf[(size_t)b * N + i] = h;
}

// ---------------- keep flags + per-128 block counts ----------------
__global__ void k_keep(const int* hbuf, const int* keys, const int* vals,
                       unsigned char* keep, int* blocksums, int cap, int N) {
  int b = blockIdx.y;
  int i = blockIdx.x * K3BLK + threadIdx.x;
  int h = hbuf[(size_t)b * N + i];
  int kp = 0;
  if (h != HSENT) {
    const int* K = keys + (size_t)b * cap;
    const int* V = vals + (size_t)b * cap;
    unsigned slot = hmix(h) & (unsigned)(cap - 1);
    for (int t = 0; t < cap; ++t) {
      int k = K[slot];
      if (k == h) { kp = (V[slot] == i); break; }
      if (k == -1) break;
      slot = (slot + 1) & (unsigned)(cap - 1);
    }
  }
  keep[(size_t)b * N + i] = (unsigned char)kp;
  unsigned long long bal = __ballot(kp);
  int lane = threadIdx.x & 31, wid = threadIdx.x >> 5;
  __shared__ int wsum[4];
  if (lane == 0) wsum[wid] = __popcll(bal);
  __syncthreads();
  if (threadIdx.x == 0)
    blocksums[b * (N / K3BLK) + blockIdx.x] = wsum[0] + wsum[1] + wsum[2] + wsum[3];
}

// ---------------- WMMA hierarchical exclusive scan of block sums ----------------
// One wave32 per batch. Each 256-chunk: P = V*U (row prefixes), S = Lstrict*T + P.
// All operands are small integers (<=2048) => exact in f16; f32 accumulate exact.
__global__ void k_scan_wmma(int* blocksums, int nblocks) {
  int b = blockIdx.x;
  int L = threadIdx.x;                 // 0..31
  bool lo = L < 16;
  int r = lo ? L : L - 16;
  __shared__ float vbuf[256];
  __shared__ float ibuf[256];
  __shared__ float tbuf[16];
  int* bs = blocksums + (size_t)b * nblocks;
  float carry = 0.0f;
  for (int base = 0; base < nblocks; base += 256) {
    for (int k = L; k < 256; k += 32) vbuf[k] = (float)bs[base + k];
    __syncthreads();
    // A1 = V (16x16 in K=0..15 of a 16x32 f16 A); B1 = U upper-triangular ones
    v16h A1, B1;
    for (int j = 0; j < 8; ++j)
      A1[j] = (_Float16)vbuf[16 * r + (lo ? j : 8 + j)];
    for (int j = 8; j < 16; ++j) A1[j] = (_Float16)0.0f;
    for (int j = 0; j < 16; ++j)
      B1[j] = (j < 8 && (lo ? j : 8 + j) <= r) ? (_Float16)1.0f : (_Float16)0.0f;
    v8f c0 = {};
    v8f P = __builtin_amdgcn_wmma_f32_16x16x32_f16(false, A1, false, B1,
                                                   (short)0, c0, false, false);
    // row totals T[m] = P[m][15] live in lanes 15 (m=0..7) and 31 (m=8..15)
    if (L == 15) { for (int g = 0; g < 8; ++g) tbuf[g] = P[g]; }
    if (L == 31) { for (int g = 0; g < 8; ++g) tbuf[8 + g] = P[g]; }
    __syncthreads();
    // A2 = strictly-lower-triangular ones; B2[k][c] = T[k]
    v16h A2, B2;
    for (int j = 0; j < 8; ++j)
      A2[j] = (lo ? (j < L) : ((8 + j) < (L - 16))) ? (_Float16)1.0f : (_Float16)0.0f;
    for (int j = 8; j < 16; ++j) A2[j] = (_Float16)0.0f;
    for (int j = 0; j < 16; ++j)
      B2[j] = (j < 8) ? (_Float16)tbuf[lo ? j : 8 + j] : (_Float16)0.0f;
    v8f S = __builtin_amdgcn_wmma_f32_16x16x32_f16(false, A2, false, B2,
                                                   (short)0, P, false, false);
    // scatter inclusive results s[16*m + n] to LDS
    for (int g = 0; g < 8; ++g) {
      int idx = lo ? (16 * g + L) : (16 * (g + 8) + (L - 16));
      ibuf[idx] = S[g];
    }
    __syncthreads();
    float tot = ibuf[255];
    for (int k = L; k < 256; k += 32) {
      float excl = carry + ibuf[k] - vbuf[k];
      bs[base + k] = (int)(excl + 0.5f);   // exact integers < 2^24
    }
    carry += tot;
    __syncthreads();
  }
}

// ---------------- order-preserving scatter ----------------
__global__ void k_scatter(const float* pcds, const int* qmm, const unsigned char* keep,
                          const int* blockoffs, float* out, int N, int B) {
  int b = blockIdx.y;
  int blk = blockIdx.x;
  int t = threadIdx.x;
  int i = blk * K3BLK + t;
  int kp = keep[(size_t)b * N + i];
  unsigned long long bal = __ballot(kp);
  int lane = t & 31, wid = t >> 5;
  __shared__ int wsum[4];
  if (lane == 0) wsum[wid] = __popcll(bal);
  __syncthreads();
  int off = blockoffs[b * (N / K3BLK) + blk];
  for (int w = 0; w < wid; ++w) off += wsum[w];
  off += __popcll(bal & ((1ull << lane) - 1ull));
  if (kp) {
    const float* p = pcds + ((size_t)b * N + i) * 3;
    float px = p[0], py = p[1], pz = p[2];
    int ic0 = (int)rintf(px / VOX) - qmm[b * 6 + 0];
    int ic1 = (int)rintf(py / VOX) - qmm[b * 6 + 1];
    int ic2 = (int)rintf(pz / VOX) - qmm[b * 6 + 2];
    size_t row = (size_t)b * N + off;
    out[row * 4 + 0] = px; out[row * 4 + 1] = py;
    out[row * 4 + 2] = pz; out[row * 4 + 3] = -1.0f;
    float* coords = out + (size_t)B * N * 4;
    coords[row * 3 + 0] = (float)ic0;
    coords[row * 3 + 1] = (float)ic1;
    coords[row * 3 + 2] = (float)ic2;
    float* mask = out + (size_t)B * N * 7;
    mask[row] = 1.0f;
  }
}

extern "C" void kernel_launch(void* const* d_in, const int* in_sizes, int n_in,
                              void* d_out, int out_size, void* d_ws, size_t ws_size,
                              hipStream_t stream) {
  const float* pcds = (const float*)d_in[0];
  const int N = NPTS;
  int B = in_sizes[0] / (N * 3);
  if (B <= 0) B = 1;
  const int NBLK = N / K3BLK;            // 2048
  float* out = (float*)d_out;

  // ---- workspace carve-out (deterministic, host-side) ----
  char* ws = (char*)d_ws;
  int* qmm = (int*)ws;                       ws += 256;
  int* hbuf = (int*)ws;                      ws += (size_t)B * N * sizeof(int);
  unsigned char* keep = (unsigned char*)ws;  ws += (size_t)B * N;
  int* blocksums = (int*)ws;                 ws += (size_t)B * NBLK * sizeof(int);
  uintptr_t a = (uintptr_t)ws; a = (a + 255) & ~(uintptr_t)255; ws = (char*)a;
  size_t used = (size_t)(ws - (char*)d_ws);
  size_t rem = (ws_size > used) ? (ws_size - used) : 0;
  size_t entPerBatch = rem / ((size_t)B * 8);
  int cap = 1 << 14;
  while (((size_t)cap << 1) <= entPerBatch && cap < (1 << 20)) cap <<= 1;
  int* keys = (int*)ws;
  int* vals = keys + (size_t)B * cap;

  k_zero_f32<<<2048, 256, 0, stream>>>(out, (size_t)B * N * 8);
  k_init<<<4096, 256, 0, stream>>>(keys, vals, (size_t)B * cap, qmm, B);
  k_minmax<<<dim3(N / 256, B), 256, 0, stream>>>(pcds, qmm, N);
  k_hash_insert<<<dim3(N / 256, B), 256, 0, stream>>>(pcds, qmm, hbuf, keys, vals, cap, N);
  k_keep<<<dim3(NBLK, B), K3BLK, 0, stream>>>(hbuf, keys, vals, keep, blocksums, cap, N);
  k_scan_wmma<<<B, 32, 0, stream>>>(blocksums, NBLK);
  k_scatter<<<dim3(NBLK, B), K3BLK, 0, stream>>>(pcds, qmm, keep, blocksums, out, N, B);
}
